// PreActResNet_69303592288764
// MI455X (gfx1250) — matrix operand order repaired and checked by reading, not compiled, not run on hardware
//
#include <hip/hip_runtime.h>
#include <hip/hip_fp16.h>

typedef __attribute__((ext_vector_type(16))) _Float16 v16h;
typedef __attribute__((ext_vector_type(8)))  float    v8f;

struct alignas(16) H8 { _Float16 h[8]; };

// ---------------------------------------------------------------------------
// Per-layer |w| mean (DoReFa E). One block, tiny layers (<=36864 elems).
// ---------------------------------------------------------------------------
__global__ __launch_bounds__(256)
void k_absmean(const float* __restrict__ w, int count, float* __restrict__ E) {
    __shared__ float s[256];
    float acc = 0.f;
    for (int i = threadIdx.x; i < count; i += 256) acc += fabsf(w[i]);
    s[threadIdx.x] = acc;
    __syncthreads();
    for (int off = 128; off > 0; off >>= 1) {
        if (threadIdx.x < off) s[threadIdx.x] += s[threadIdx.x + off];
        __syncthreads();
    }
    if (threadIdx.x == 0) *E = s[0] / (float)count;
}

// ---------------------------------------------------------------------------
// Quantize weights to sign(w)*E (f16) and pack straight into WMMA B-fragment
// order: [ctile][kstep][lane][16 halves];  lane<16 -> K rows {0..7,16..23} of the
// kstep, lane>=16 -> K rows {8..15,24..31};  N (=cout) = lane&15.  K index =
// (ky*ks+kx)*Cin + ci  (matches the NHWC im2col gather on the A side).
// ---------------------------------------------------------------------------
__global__ __launch_bounds__(256)
void k_pack_w(const float* __restrict__ W /*OIHW*/, const float* __restrict__ E,
              _Float16* __restrict__ Wf, int Cin, int Cout, int ksize,
              int K, int nks, int nct) {
    int t = blockIdx.x * 256 + threadIdx.x;
    int total = nct * nks * 32;
    if (t >= total) return;
    int lane  = t & 31;
    int ks    = (t >> 5) % nks;
    int ctile = t / (nks * 32);
    int hi = lane >> 4, ln = lane & 15;
    int co = ctile * 16 + ln;
    float e = *E;
    _Float16* out = Wf + ((size_t)(ctile * nks + ks) * 32 + lane) * 16;
    for (int i = 0; i < 16; ++i) {
        int k = (ks << 5) + ((i >> 3) << 4) + (hi << 3) + (i & 7);
        float v = 0.f;
        if (k < K) {
            int cell = k / Cin, ci = k - cell * Cin;
            int ky = cell / ksize, kx = cell - ky * ksize;
            float wv = W[(((size_t)co * Cin + ci) * ksize + ky) * ksize + kx];
            v = (wv > 0.f) ? e : (wv < 0.f ? -e : 0.f);   // jnp.sign semantics
        }
        out[i] = (_Float16)v;
    }
}

// ---------------------------------------------------------------------------
// NCHW f32 -> NHWC f16 (first conv input)
// ---------------------------------------------------------------------------
__global__ __launch_bounds__(256)
void k_nchw_to_nhwc(const float* __restrict__ X, _Float16* __restrict__ A,
                    int N, int C, int H, int W) {
    long i = (long)blockIdx.x * 256 + threadIdx.x;
    long total = (long)N * C * H * W;
    if (i >= total) return;
    int c = (int)(i % C); long t = i / C;
    int w = (int)(t % W); t /= W;
    int h = (int)(t % H); int n = (int)(t / H);
    A[i] = (_Float16)X[(((long)n * C + c) * H + h) * W + w];
}

// ---------------------------------------------------------------------------
// Per-channel BN stats (training-mode, biased var) -> scale/shift.
// X: [P][C] NHWC f32.  stats[c]=g*rsqrt(var+eps), stats[C+c]=b-mu*scale.
// ---------------------------------------------------------------------------
__global__ __launch_bounds__(256)
void k_bn_stats(const float* __restrict__ X, const float* __restrict__ g,
                const float* __restrict__ b, float* __restrict__ stats,
                int C, long P) {
    __shared__ float s1[256], s2[256];
    int c = blockIdx.x;
    float sum = 0.f, sq = 0.f;
    for (long p = threadIdx.x; p < P; p += 256) {
        float v = X[p * C + c];
        sum += v; sq += v * v;
    }
    s1[threadIdx.x] = sum; s2[threadIdx.x] = sq;
    __syncthreads();
    for (int off = 128; off > 0; off >>= 1) {
        if (threadIdx.x < off) {
            s1[threadIdx.x] += s1[threadIdx.x + off];
            s2[threadIdx.x] += s2[threadIdx.x + off];
        }
        __syncthreads();
    }
    if (threadIdx.x == 0) {
        float inv = 1.f / (float)P;
        float mu  = s1[0] * inv;
        float var = s2[0] * inv - mu * mu;
        float sc  = g[c] * rsqrtf(var + 1e-5f);
        stats[c]     = sc;
        stats[C + c] = b[c] - mu * sc;
    }
}

// ---------------------------------------------------------------------------
// Fused BN apply + ReLU + f16 quantize (produces next GEMM A operand)
// ---------------------------------------------------------------------------
__global__ __launch_bounds__(256)
void k_bn_relu(const float* __restrict__ X, const float* __restrict__ stats,
               _Float16* __restrict__ A, int C, long total) {
    long i = (long)blockIdx.x * 256 + threadIdx.x;
    if (i >= total) return;
    int c = (int)(i % C);
    float v = fmaxf(X[i] * stats[c] + stats[C + c], 0.f);
    A[i] = (_Float16)v;
}

// ---------------------------------------------------------------------------
// Implicit-GEMM conv via v_wmma_f32_16x16x32_f16, fully specialized.
// One wave = 16 output pixels x (16*NCT = all) output channels; the K loop
// (NKS steps of 32) is constexpr and fully unrolled, so all im2col index
// math, K-pad zero-fills, and weight-fragment offsets constant-fold.
// A gather is reused across NCT accumulators (arithmetic intensity x NCT).
// Exact grid -> no early exit -> EXEC all-ones at every WMMA (ISA req).
// ---------------------------------------------------------------------------
template<int CIN, int KS, int NCT>
__global__ __launch_bounds__(256)
void k_conv_wmma(const _Float16* __restrict__ A,
                 const _Float16* __restrict__ Wf,
                 const float* __restrict__ resid,
                 float* __restrict__ Out,
                 int H, int W, int stride, int pad,
                 int pixShift, int woShift) {
    constexpr int K    = CIN * KS * KS;
    constexpr int NKS  = (K + 31) / 32;
    constexpr int COUT = NCT * 16;

    int wave = (blockIdx.x * 256 + threadIdx.x) >> 5;   // == M-tile index
    int lane = threadIdx.x & 31;
    int hi = lane >> 4;
    int ln = lane & 15;

    // A-row pixel for this lane (all dims are powers of two -> shifts only)
    int m  = (wave << 4) + ln;
    int n  = m >> pixShift;
    int r  = m & ((1 << pixShift) - 1);
    int ho = r >> woShift;
    int wo = r & ((1 << woShift) - 1);
    int hbase = ho * stride - pad;
    int wbase = wo * stride - pad;
    const _Float16* abase = A + (ptrdiff_t)((n * H + hbase) * W + wbase) * CIN;

    const _Float16* wfl = Wf + lane * 16;

    v8f acc[NCT] = {};

#pragma unroll
    for (int ks = 0; ks < NKS; ++ks) {
        // ---- A fragment: lane<16 holds K {0..7,16..23}, lane>=16 {8..15,24..31}
        v16h a;
#pragma unroll
        for (int g = 0; g < 2; ++g) {
            const int k0c = (ks << 5) + (g << 4);       // 16-aligned, constant
            if constexpr ((CIN & 7) == 0) {
                // K is a multiple of 16 here, so both hi-halves of a group are
                // jointly valid or jointly K-pad; +8 never crosses a (ky,kx) cell.
                if (k0c + 16 <= K) {
                    int cell = k0c / CIN;               // folds to a constant
                    int ci   = (k0c - cell * CIN) + (hi << 3);
                    int ky   = cell / KS, kx = cell - ky * KS;
                    int hh = hbase + ky, ww = wbase + kx;
                    if (hh >= 0 && hh < H && ww >= 0 && ww < W) {
                        const H8* p = (const H8*)(abase + (ky * W + kx) * CIN + ci);
                        H8 v = *p;                      // global_load_b128
#pragma unroll
                        for (int j = 0; j < 8; ++j) a[g * 8 + j] = v.h[j];
                    } else {
#pragma unroll
                        for (int j = 0; j < 8; ++j) a[g * 8 + j] = (_Float16)0.f;
                    }
                } else {                                // K-pad tail -> zeros
#pragma unroll
                    for (int j = 0; j < 8; ++j) a[g * 8 + j] = (_Float16)0.f;
                }
            } else {                                    // Cin=3 stem (NKS==1)
#pragma unroll
                for (int j = 0; j < 8; ++j) {
                    int k = k0c + (hi << 3) + j;
                    _Float16 val = (_Float16)0.f;
                    if (k < K) {
                        int cell = k / CIN, ci = k - cell * CIN;
                        int ky = cell / KS, kx = cell - ky * KS;
                        int hh = hbase + ky, ww = wbase + kx;
                        if (hh >= 0 && hh < H && ww >= 0 && ww < W)
                            val = abase[(ky * W + kx) * CIN + ci];
                    }
                    a[g * 8 + j] = val;
                }
            }
        }

        // ---- one WMMA per cout-tile, reusing the A fragment
#pragma unroll
        for (int ct = 0; ct < NCT; ++ct) {
            v16h b;
            const H8* pb = (const H8*)(wfl + (size_t)(ct * NKS + ks) * 512);
            H8 b0 = pb[0], b1 = pb[1];                  // two 16B loads, const offs
#pragma unroll
            for (int j = 0; j < 8; ++j) { b[j] = b0.h[j]; b[8 + j] = b1.h[j]; }
            acc[ct] = __builtin_amdgcn_wmma_f32_16x16x32_f16(
                false, a, false, b, (short)0, acc[ct], false, false);
        }
    }

    // ---- epilogue: c[r] -> M = r + 8*hi, N = ln;  Out is [pixel][COUT]
#pragma unroll
    for (int ct = 0; ct < NCT; ++ct) {
        int co = (ct << 4) + ln;
#pragma unroll
        for (int rI = 0; rI < 8; ++rI) {
            int mm = (wave << 4) + rI + (hi << 3);
            size_t oidx = (size_t)mm * COUT + co;
            float v = acc[ct][rI];
            if (resid) v += resid[oidx];
            Out[oidx] = v;
        }
    }
}

// ---------------------------------------------------------------------------
// Head: BN-fold + global average pool + 64x10 logits. One block per image.
// X: [n][64 pixels][64 ch] f32.
// ---------------------------------------------------------------------------
__global__ __launch_bounds__(256)
void k_head(const float* __restrict__ X, const float* __restrict__ stats,
            const float* __restrict__ lw, const float* __restrict__ lb,
            float* __restrict__ out) {
    __shared__ float part[256];
    __shared__ float pooled[64];
    int n = blockIdx.x;
    int c = threadIdx.x & 63;
    int q = threadIdx.x >> 6;        // 4 partial sums per channel
    float s = 0.f;
    for (int p = q; p < 64; p += 4) s += X[((size_t)n * 64 + p) * 64 + c];
    part[threadIdx.x] = s;
    __syncthreads();
    if (q == 0) {
        float m = (part[c] + part[64 + c] + part[128 + c] + part[192 + c]) * (1.f / 64.f);
        pooled[c] = m * stats[c] + stats[64 + c];
    }
    __syncthreads();
    if (threadIdx.x < 10) {
        float acc = lb[threadIdx.x];
        for (int k = 0; k < 64; ++k) acc += pooled[k] * lw[threadIdx.x * 64 + k];
        out[(size_t)n * 10 + threadIdx.x] = acc;
    }
}

// ---------------------------------------------------------------------------
static void launch_conv(hipStream_t stream, const _Float16* A, const _Float16* Wf,
                        const float* resid, float* Out,
                        int N, int H, int W, int Cin, int Ho, int Wo, int Cout,
                        int ks, int st, int pad) {
    long waves = (long)N * Ho * Wo / 16;      // one wave owns 16 pixels x all Cout
    int blocks = (int)(waves / 8);            // 256 threads = 8 wave32
    int pixShift = __builtin_ctz(Ho * Wo);
    int woShift  = __builtin_ctz(Wo);
#define LC(CI, KK, NC) \
    k_conv_wmma<CI, KK, NC><<<blocks, 256, 0, stream>>>(A, Wf, resid, Out, H, W, st, pad, pixShift, woShift)
    if      (Cin == 3)                           LC(3, 3, 1);
    else if (Cin == 16 && ks == 3 && Cout == 16) LC(16, 3, 1);
    else if (Cin == 16 && ks == 3)               LC(16, 3, 2);
    else if (Cin == 16 && ks == 1)               LC(16, 1, 2);
    else if (Cin == 32 && ks == 3 && Cout == 32) LC(32, 3, 2);
    else if (Cin == 32 && ks == 3)               LC(32, 3, 4);
    else if (Cin == 32 && ks == 1)               LC(32, 1, 4);
    else                                         LC(64, 3, 4);
#undef LC
}

extern "C" void kernel_launch(void* const* d_in, const int* in_sizes, int n_in,
                              void* d_out, int out_size, void* d_ws, size_t ws_size,
                              hipStream_t stream) {
    (void)in_sizes; (void)n_in; (void)out_size; (void)ws_size;
    const int N = 1024;

    // ---- workspace carve-out (everything L2-sized; 256B aligned)
    char* ws = (char*)d_ws;
    size_t off = 0;
    auto alloc = [&](size_t bytes) -> void* {
        void* p = (void*)(ws + off);
        off = (off + bytes + 255) & ~(size_t)255;
        return p;
    };
    float*    buf_x = (float*)alloc(67108864);      // block input / residual (f32 NHWC)
    float*    buf_y = (float*)alloc(67108864);      // conv0 output
    float*    buf_s = (float*)alloc(33554432);      // skip-conv output
    _Float16* a0    = (_Float16*)alloc(33554432);   // relu(bn(x)) f16
    _Float16* a1    = (_Float16*)alloc(33554432);   // relu(bn(y)) f16
    _Float16* wfrag = (_Float16*)alloc(1u << 20);   // all packed weights (~0.55MB)
    float*    dE    = (float*)alloc(32 * sizeof(float));
    float*    stats = (float*)alloc(128 * sizeof(float));

    // ---- enumerate inputs (setup_inputs dict insertion order) + conv table
    struct CD { const float* w; int cin, cout, ks; size_t foff; };
    CD cd[21]; int ncd = 0; size_t fo = 0;
    auto addconv = [&](const float* w, int cin, int cout, int ks) -> int {
        int K = cin * ks * ks, nks = (K + 31) / 32, nct = cout / 16;
        cd[ncd].w = w; cd[ncd].cin = cin; cd[ncd].cout = cout;
        cd[ncd].ks = ks; cd[ncd].foff = fo;
        fo += (size_t)nct * nks * 512;
        return ncd++;
    };

    const float* x = (const float*)d_in[0];
    int ii = 1;
    int firstConv = addconv((const float*)d_in[ii++], 3, 16, 3);

    struct BlkRef { const float *bn0g, *bn0b, *bn1g, *bn1b; int c0, c1, sk, cout, stride; };
    BlkRef blk[9];
    const int planes[3] = {16, 32, 64}, strides[3] = {1, 2, 2};
    int cin = 16, bi = 0;
    for (int g = 0; g < 3; ++g)
        for (int u = 0; u < 3; ++u) {
            int s = (u == 0) ? strides[g] : 1;
            int cout = planes[g];
            BlkRef& B = blk[bi];
            B.bn0g = (const float*)d_in[ii++];
            B.bn0b = (const float*)d_in[ii++];
            B.c0   = addconv((const float*)d_in[ii++], cin, cout, 3);
            B.bn1g = (const float*)d_in[ii++];
            B.bn1b = (const float*)d_in[ii++];
            B.c1   = addconv((const float*)d_in[ii++], cout, cout, 3);
            B.sk   = (s != 1 || cin != cout) ? addconv((const float*)d_in[ii++], cin, cout, 1) : -1;
            B.cout = cout; B.stride = s;
            cin = cout; ++bi;
        }
    const float* bn_g = (const float*)d_in[ii++];
    const float* bn_b = (const float*)d_in[ii++];
    const float* lw   = (const float*)d_in[ii++];
    const float* lb   = (const float*)d_in[ii++];

    // ---- quantize + fragment-pack every conv layer's weights
    for (int i = 0; i < ncd; ++i) {
        int K = cd[i].cin * cd[i].ks * cd[i].ks;
        k_absmean<<<1, 256, 0, stream>>>(cd[i].w, cd[i].cout * K, dE + i);
        int nks = (K + 31) / 32, nct = cd[i].cout / 16;
        int tot = nct * nks * 32;
        k_pack_w<<<(tot + 255) / 256, 256, 0, stream>>>(
            cd[i].w, dE + i, wfrag + cd[i].foff, cd[i].cin, cd[i].cout, cd[i].ks, K, nks, nct);
    }

    // ---- stem: NCHW->NHWC f16, then conv0 (no BN before it)
    {
        long tot = (long)N * 3 * 32 * 32;
        k_nchw_to_nhwc<<<(int)((tot + 255) / 256), 256, 0, stream>>>(x, a0, N, 3, 32, 32);
    }
    launch_conv(stream, a0, wfrag + cd[firstConv].foff, nullptr, buf_x,
                N, 32, 32, 3, 32, 32, 16, 3, 1, 1);

    // ---- pre-activation residual blocks
    int H = 32, W = 32, C = 16;
    for (int b = 0; b < 9; ++b) {
        BlkRef& B = blk[b];
        int Ho = H / B.stride, Wo = W / B.stride;
        long Pin  = (long)N * H * W;
        long Pout = (long)N * Ho * Wo;

        k_bn_stats<<<C, 256, 0, stream>>>(buf_x, B.bn0g, B.bn0b, stats, C, Pin);
        k_bn_relu<<<(int)((Pin * C + 255) / 256), 256, 0, stream>>>(buf_x, stats, a0, C, Pin * C);

        launch_conv(stream, a0, wfrag + cd[B.c0].foff, nullptr, buf_y,
                    N, H, W, C, Ho, Wo, B.cout, 3, B.stride, 1);

        const float* resid = buf_x;
        if (B.sk >= 0) {
            launch_conv(stream, a0, wfrag + cd[B.sk].foff, nullptr, buf_s,
                        N, H, W, C, Ho, Wo, B.cout, 1, B.stride, 0);
            resid = buf_s;
        }

        k_bn_stats<<<B.cout, 256, 0, stream>>>(buf_y, B.bn1g, B.bn1b, stats, B.cout, Pout);
        k_bn_relu<<<(int)((Pout * B.cout + 255) / 256), 256, 0, stream>>>(
            buf_y, stats, a1, B.cout, Pout * B.cout);

        launch_conv(stream, a1, wfrag + cd[B.c1].foff, resid, buf_x,
                    N, Ho, Wo, B.cout, Ho, Wo, B.cout, 3, 1, 1);

        H = Ho; W = Wo; C = B.cout;
    }

    // ---- head: final BN + global average pool + logits
    k_bn_stats<<<64, 256, 0, stream>>>(buf_x, bn_g, bn_b, stats, 64, (long)N * 64);
    k_head<<<N, 256, 0, stream>>>(buf_x, stats, lw, lb, (float*)d_out);
}
